// PointNetSetAbstractionMsg_81303730913469
// MI455X (gfx1250) — matrix operand order repaired and checked by reading, not compiled, not run on hardware
//
#include <hip/hip_runtime.h>

#define BB 8
#define NN 4096
#define DD 64
#define SS 1024
#define OUTC 320
#define NEWXYZ_ELEMS (BB*SS*3)

typedef __attribute__((ext_vector_type(8)))  _Float16 v8h;
typedef __attribute__((ext_vector_type(16))) _Float16 v16h;
typedef __attribute__((ext_vector_type(8)))  float    v8f;

// ---------------------------------------------------------------------------
// Kernel 1: Farthest Point Sampling. One block per batch; points + running
// distances live in registers (16 pts/thread), argmax via LDS tree reduce.
// Writes new_xyz to d_out (first tuple element) and to workspace.
// ---------------------------------------------------------------------------
__global__ __launch_bounds__(256)
void fps_kernel(const float* __restrict__ xyz, float* __restrict__ out,
                float* __restrict__ newxyz_ws)
{
  __shared__ float rv[256];
  __shared__ int   ri[256];
  const int b   = blockIdx.x;
  const int tid = threadIdx.x;
  const float* xb = xyz + (size_t)b * NN * 3;
  float px[16], py[16], pz[16], dist[16];
  const int base = tid * 16;
  for (int j = 0; j < 16; ++j) {
    px[j] = xb[(base + j) * 3 + 0];
    py[j] = xb[(base + j) * 3 + 1];
    pz[j] = xb[(base + j) * 3 + 2];
    dist[j] = 1e10f;
  }
  int cur = 0;
  for (int i = 0; i < SS; ++i) {
    const float cx = xb[cur * 3 + 0], cy = xb[cur * 3 + 1], cz = xb[cur * 3 + 2];
    if (tid == 0) {
      float* o = out + ((size_t)b * SS + i) * 3;
      o[0] = cx; o[1] = cy; o[2] = cz;
      float* w = newxyz_ws + ((size_t)b * SS + i) * 3;
      w[0] = cx; w[1] = cy; w[2] = cz;
    }
    float best = -1.0f; int bi = base;
    for (int j = 0; j < 16; ++j) {
      const float dx = px[j] - cx, dy = py[j] - cy, dz = pz[j] - cz;
      const float d = dx * dx + dy * dy + dz * dz;
      dist[j] = fminf(dist[j], d);
      if (dist[j] > best) { best = dist[j]; bi = base + j; }
    }
    rv[tid] = best; ri[tid] = bi;
    __syncthreads();
    for (int s2 = 128; s2 > 0; s2 >>= 1) {
      if (tid < s2) {
        const float ov = rv[tid + s2]; const int oi = ri[tid + s2];
        if (ov > rv[tid] || (ov == rv[tid] && oi < ri[tid])) { rv[tid] = ov; ri[tid] = oi; }
      }
      __syncthreads();
    }
    cur = ri[0];
    __syncthreads();
  }
}

// ---------------------------------------------------------------------------
// Kernel 2: fused 3-radius ball query. One wave32 per centroid; ordered
// ballot scan over the 4096 points produces the first-k in-radius indices
// for all three scales in one pass (matches jnp.sort-of-masked-index trick).
// ---------------------------------------------------------------------------
__global__ __launch_bounds__(256)
void ballquery_kernel(const float* __restrict__ xyz, const float* __restrict__ newxyz,
                      int* __restrict__ i0, int* __restrict__ i1, int* __restrict__ i2)
{
  const int tid = threadIdx.x, lane = tid & 31, wave = tid >> 5;
  const int c = blockIdx.x * 8 + wave;
  const int b = c >> 10;                       // c / SS
  const float cx = newxyz[c * 3 + 0], cy = newxyz[c * 3 + 1], cz = newxyz[c * 3 + 2];
  const float* xb = xyz + (size_t)b * NN * 3;
  int* p0 = i0 + (size_t)c * 16;
  int* p1 = i1 + (size_t)c * 32;
  int* p2 = i2 + (size_t)c * 128;
  int c0 = 0, c1 = 0, c2 = 0;
  const float r0 = 0.01f, r1 = 0.04f, r2 = 0.16f;   // radius^2
  const unsigned below = (1u << lane) - 1u;
  for (int n0 = 0; n0 < NN; n0 += 32) {
    const int p = n0 + lane;
    const float dx = xb[p * 3 + 0] - cx;
    const float dy = xb[p * 3 + 1] - cy;
    const float dz = xb[p * 3 + 2] - cz;
    const float d = dx * dx + dy * dy + dz * dz;
    const unsigned m0 = (unsigned)__ballot(d <= r0);
    if (d <= r0) { const int pos = c0 + __popc(m0 & below); if (pos < 16)  p0[pos] = p; }
    c0 += __popc(m0);
    const unsigned m1 = (unsigned)__ballot(d <= r1);
    if (d <= r1) { const int pos = c1 + __popc(m1 & below); if (pos < 32)  p1[pos] = p; }
    c1 += __popc(m1);
    const unsigned m2 = (unsigned)__ballot(d <= r2);
    if (d <= r2) { const int pos = c2 + __popc(m2 & below); if (pos < 128) p2[pos] = p; }
    c2 += __popc(m2);
    if (c0 >= 16 && c1 >= 32 && c2 >= 128) break;
  }
  // pad with first neighbor (centroid itself guarantees >=1 per scale)
  const int f0 = p0[0], f1 = p1[0], f2 = p2[0];
  for (int pos = c0 + lane; pos < 16;  pos += 32) p0[pos] = f0;
  for (int pos = c1 + lane; pos < 32;  pos += 32) p1[pos] = f1;
  for (int pos = c2 + lane; pos < 128; pos += 32) p2[pos] = f2;
}

// ---------------------------------------------------------------------------
// WMMA helpers + fused gather -> MLP(3) -> maxpool kernel (per scale).
// 128 gathered rows per workgroup; 8 waves each own one 16-row M tile.
// ---------------------------------------------------------------------------
__device__ __forceinline__ v16h ld16(const _Float16* p0, const _Float16* p1)
{
  const v8h lo = *(const v8h*)p0;
  const v8h hi = *(const v8h*)p1;
  return __builtin_shufflevector(lo, hi, 0,1,2,3,4,5,6,7,8,9,10,11,12,13,14,15);
}

__device__ void mlp_layer(const _Float16* __restrict__ actIn, int strideIn, int Kdim,
                          _Float16* __restrict__ actOut, int strideOut, int Ndim,
                          const float* __restrict__ Wg, int cinAct,
                          const float* __restrict__ Bg,
                          _Float16* __restrict__ sW, float* __restrict__ sBias,
                          unsigned* __restrict__ sPool, int C3dim, int Kpts, bool pool)
{
  const int tid = threadIdx.x;
  const int wave = tid >> 5, lane = tid & 31, lrow = lane & 15, lhalf = lane >> 4;
  const int mbase = wave * 16;
  for (int o = tid; o < Ndim; o += 256) sBias[o] = Bg[o];
  for (int nb = 0; nb < Ndim; nb += 64) {
    const int NB = (Ndim - nb) < 64 ? (Ndim - nb) : 64;
    // stage W (f16) for this N block: sW[n*Kdim + k] = W[nb+n, k], zero-pad k>=cinAct
    for (int e = tid; e < Kdim * NB; e += 256) {
      const int cc = e / NB, oo = e - cc * NB;
      const float v = (cc < cinAct) ? Wg[(size_t)(nb + oo) * cinAct + cc] : 0.0f;
      sW[oo * Kdim + cc] = (_Float16)v;
    }
    __syncthreads();
    for (int nt = 0; nt < NB / 16; ++nt) {
      v8f acc = {};
      for (int kt = 0; kt < Kdim / 32; ++kt) {
        // A fragment: row = mbase+lrow; K chunks {base..base+7, base+16..base+23}
        const _Float16* pa = actIn + (mbase + lrow) * strideIn + kt * 32 + lhalf * 8;
        const v16h a = ld16(pa, pa + 16);
        // B fragment: col = nt*16+lrow; K chunks {base..base+7, base+16..base+23}
        const _Float16* pb = sW + (nt * 16 + lrow) * Kdim + kt * 32 + lhalf * 8;
        const v16h bm = ld16(pb, pb + 16);
        acc = __builtin_amdgcn_wmma_f32_16x16x32_f16(false, a, false, bm,
                                                     (short)0, acc, false, false);
      }
      const int col = nb + nt * 16 + lrow;
      const float bias = sBias[col];
      if (!pool) {
        for (int j = 0; j < 8; ++j) {
          const int row = mbase + j + (lhalf << 3);     // C/D layout: VGPR j -> M=j(+8)
          float v = acc[j] + bias; v = v > 0.0f ? v : 0.0f;
          actOut[row * strideOut + col] = (_Float16)v;
        }
      } else {
        for (int j = 0; j < 8; ++j) {
          const int row = mbase + j + (lhalf << 3);
          float v = acc[j] + bias; v = v > 0.0f ? v : 0.0f;
          const int g = row / Kpts;
          atomicMax(&sPool[g * C3dim + col], __float_as_uint(v));  // ds_max_u32; valid: v>=0
        }
      }
    }
    __syncthreads();
  }
}

template<int K, int C1, int C2, int C3, int OUT_OFF>
__global__ __launch_bounds__(256)
void msg_mlp_kernel(const float* __restrict__ xyz, const float* __restrict__ feat,
                    const float* __restrict__ newxyz, const int* __restrict__ idx,
                    const float* __restrict__ w1, const float* __restrict__ b1,
                    const float* __restrict__ w2, const float* __restrict__ b2,
                    const float* __restrict__ w3, const float* __restrict__ b3,
                    float* __restrict__ out)
{
  constexpr int ROWS = 128, CINP = 96, CIN = 67, G = ROWS / K;
  __shared__ _Float16 sA[ROWS * CINP];   // input (stride 96) / layer-2 out (stride C2)
  __shared__ _Float16 sB[ROWS * 64];     // layer-1 out (stride C1 <= 64)
  __shared__ _Float16 sW[96 * 64];       // f16 W^T block (K<=96, Nblock<=64)
  __shared__ float    sBias[128];
  __shared__ unsigned sPool[G * C3];     // maxpool accumulators (float bits)
  const int tid = threadIdx.x;
  const int cbase = blockIdx.x * G;
  for (int e = tid; e < G * C3; e += 256) sPool[e] = 0u;
  // gather: 2 threads per row; 3 relative-xyz + 64 features + zero pad to 96
  {
    const int r = tid >> 1, h = tid & 1;
    const int c = cbase + r / K, g = r % K;
    const int b = c >> 10;
    const int p = idx[(size_t)c * K + g];
    const float* px = xyz + ((size_t)b * NN + p) * 3;
    const float* pf = feat + ((size_t)b * NN + p) * DD;
    const float* pc = newxyz + (size_t)c * 3;
    for (int col = h * 48; col < h * 48 + 48; ++col) {
      float v;
      if (col < 3)        v = px[col] - pc[col];
      else if (col < CIN) v = pf[col - 3];
      else                v = 0.0f;
      sA[r * CINP + col] = (_Float16)v;
    }
  }
  __builtin_prefetch(w2, 0, 1);   // global_prefetch_b8: warm later-layer weights
  __builtin_prefetch(w3, 0, 1);
  __syncthreads();
  mlp_layer(sA, CINP, CINP, sB, C1, C1, w1, CIN, b1, sW, sBias, sPool, C3, K, false);
  mlp_layer(sB, C1,   C1,   sA, C2, C2, w2, C1,  b2, sW, sBias, sPool, C3, K, false);
  mlp_layer(sA, C2,   C2,   nullptr, 0, C3, w3, C2, b3, sW, sBias, sPool, C3, K, true);
  __syncthreads();
  for (int e = tid; e < G * C3; e += 256) {
    const int g = e / C3, o = e - g * C3;
    const int c = cbase + g;                 // c == b*SS + s
    out[NEWXYZ_ELEMS + (size_t)c * OUTC + OUT_OFF + o] = __uint_as_float(sPool[e]);
  }
}

// ---------------------------------------------------------------------------
extern "C" void kernel_launch(void* const* d_in, const int* in_sizes, int n_in,
                              void* d_out, int out_size, void* d_ws, size_t ws_size,
                              hipStream_t stream)
{
  (void)in_sizes; (void)n_in; (void)out_size; (void)ws_size;
  const float* xyz  = (const float*)d_in[0];
  const float* feat = (const float*)d_in[1];
  const float* w[3][3]; const float* bi[3][3];
  int c = 2;
  for (int i = 0; i < 3; ++i)
    for (int j = 0; j < 3; ++j) { w[i][j] = (const float*)d_in[c++]; bi[i][j] = (const float*)d_in[c++]; }
  float* out = (float*)d_out;
  // workspace carve: new_xyz (96KB) | idx r=0.1,k=16 (512KB) | k=32 (1MB) | k=128 (4MB)
  char* ws = (char*)d_ws;
  float* newxyz = (float*)ws;
  int* i0 = (int*)(ws + 98304);
  int* i1 = (int*)(ws + 98304 + 524288);
  int* i2 = (int*)(ws + 98304 + 524288 + 1048576);

  fps_kernel<<<BB, 256, 0, stream>>>(xyz, out, newxyz);
  ballquery_kernel<<<(BB * SS) / 8, 256, 0, stream>>>(xyz, newxyz, i0, i1, i2);
  msg_mlp_kernel<16, 32, 32, 64, 0>
      <<<(BB * SS) / 8, 256, 0, stream>>>(xyz, feat, newxyz, i0,
          w[0][0], bi[0][0], w[0][1], bi[0][1], w[0][2], bi[0][2], out);
  msg_mlp_kernel<32, 64, 64, 128, 64>
      <<<(BB * SS) / 4, 256, 0, stream>>>(xyz, feat, newxyz, i1,
          w[1][0], bi[1][0], w[1][1], bi[1][1], w[1][2], bi[1][2], out);
  msg_mlp_kernel<128, 64, 96, 128, 192>
      <<<(BB * SS) / 1, 256, 0, stream>>>(xyz, feat, newxyz, i2,
          w[2][0], bi[2][0], w[2][1], bi[2][1], w[2][2], bi[2][2], out);
}
